// A3TGCN_58506044506786
// MI455X (gfx1250) — matrix operand order repaired and checked by reading, not compiled, not run on hardware
//
#include <hip/hip_runtime.h>
#include <hip/hip_bf16.h>
#include <math.h>

typedef __attribute__((ext_vector_type(16))) __bf16 v16bf;
typedef __attribute__((ext_vector_type(8)))  __bf16 v8bf;
typedef __attribute__((ext_vector_type(8)))  float  v8f;

#define KMAX 192
#define LDSP (KMAX + 8)   // 200 elems = 400 B pitch, multiple of 16 B
#define ACT_NONE 0
#define ACT_SIGMOID 1
#define ACT_TANH 2

static inline int ceil_div(int a, int b) { return (a + b - 1) / b; }

// ---------------- elementwise / setup kernels ----------------

__global__ void fill_f32_kernel(float* p, float v, int n) {
    int i = blockIdx.x * blockDim.x + threadIdx.x;
    if (i < n) p[i] = v;
}

__global__ void deg_scatter_kernel(const int* __restrict__ dst, const float* __restrict__ ea,
                                   float* __restrict__ deg, int E) {
    int e = blockIdx.x * blockDim.x + threadIdx.x;
    if (e >= E) return;
    atomicAdd(&deg[dst[e]], ea[(size_t)e * 2 + 1]);
}

__global__ void dinv_kernel(const float* __restrict__ deg, float* __restrict__ dinv, int n) {
    int i = blockIdx.x * blockDim.x + threadIdx.x;
    if (i >= n) return;
    float d = deg[i];
    dinv[i] = (d > 0.0f) ? rsqrtf(d) : 0.0f;
}

__global__ void norm_kernel(const int* __restrict__ src, const int* __restrict__ dst,
                            const float* __restrict__ ea, const float* __restrict__ dinv,
                            float* __restrict__ nrm, int E) {
    int e = blockIdx.x * blockDim.x + threadIdx.x;
    if (e >= E) return;
    nrm[e] = dinv[src[e]] * ea[(size_t)e * 2 + 1] * dinv[dst[e]];
}

// dst[r*ldd + coff + c] = bf16( trans ? src[c*R + r] : src[r*C + c] )
__global__ void pack_w_kernel(const float* __restrict__ src, __bf16* __restrict__ dst,
                              int R, int C, int ldd, int coff, int trans) {
    int i = blockIdx.x * blockDim.x + threadIdx.x;
    if (i >= R * C) return;
    int r = i / C, c = i - r * C;
    float v = trans ? src[(size_t)c * R + r] : src[(size_t)r * C + c];
    dst[(size_t)r * ldd + coff + c] = (__bf16)v;
}

__global__ void copy_bias_kernel(const float* __restrict__ s, float* __restrict__ d, int n, int off) {
    int i = blockIdx.x * blockDim.x + threadIdx.x;
    if (i < n) d[off + i] = s[i];
}

// o[n*cols + c] = bf16( xi[n*srow + c*scol + soff] )
__global__ void pack_xi_kernel(const float* __restrict__ xi, __bf16* __restrict__ o,
                               int Nn, int cols, int srow, int scol, int soff) {
    int i = blockIdx.x * blockDim.x + threadIdx.x;
    if (i >= Nn * cols) return;
    int n = i / cols, c = i - n * cols;
    o[i] = (__bf16)xi[(size_t)n * srow + (size_t)c * scol + soff];
}

// ---------------- WMMA GEMM: out = act(A(MxK,bf16) @ B(KxNcol,bf16) + bias) ----------------
// block tile 64x64, 8 waves as 4(row)x2(col-group); each wave: 16 rows x 32 cols = 2 WMMA accs.
// Requires lda == K (true for all call sites), K % 32 == 0, K <= KMAX.
__global__ __launch_bounds__(256)
void gemm_bf16_kernel(const __bf16* __restrict__ A, int lda,
                      const __bf16* __restrict__ B, int ldb,
                      const float* __restrict__ bias,
                      float* __restrict__ outF, __bf16* __restrict__ outH, int ldo,
                      int M, int K, int Ncol, int act)
{
    __shared__ __bf16 As[64][LDSP];   // As[row][k]
    __shared__ __bf16 Bt[64][LDSP];   // Bt[col][k] = B[k][col_base+col]  (transposed!)
    const int tid = threadIdx.x;
    const int row_base = blockIdx.x * 64;
    const int col_base = blockIdx.y * 64;
    const __bf16 z = (__bf16)0.0f;

    // A tile: 16-byte vector loads (K multiple of 8, rows 16B aligned)
    const int kv = K >> 3;                       // 8-elem chunks per row
    for (int i = tid; i < 64 * kv; i += 256) {
        int r = i / kv, kc = (i - r * kv) << 3;
        int gr = row_base + r;
        v8bf v = {z, z, z, z, z, z, z, z};
        if (gr < M) v = *(const v8bf*)(A + (size_t)gr * lda + kc);
        *(v8bf*)&As[r][kc] = v;
    }
    // B tile, transposed into LDS; global reads k-major (coalesced over columns)
    for (int i = tid; i < K * 64; i += 256) {
        int k = i >> 6, c = i & 63;
        int gc = col_base + c;
        Bt[c][k] = (gc < Ncol) ? B[(size_t)k * ldb + gc] : z;
    }
    __syncthreads();

    const int wid = tid >> 5, lane = tid & 31;
    const int wr  = wid & 3;        // row tile 0..3
    const int wcg = wid >> 2;       // col group 0..1 (32 cols each)
    const int l15 = lane & 15, lhi = lane >> 4;

    const int arow = wr * 16 + l15;       // A frag: lanes 0-15/16-31 both rows 0-15
    const int kgA  = lhi * 16;            // A frag: hi half-wave K+16..31
    const int bc0  = wcg * 32 + l15;      // B frag: lane -> column
    const int bc1  = bc0 + 16;
    const int kgB  = lhi * 16;            // B frag: hi half-wave K+16..31

    v8f acc0 = {0.f, 0.f, 0.f, 0.f, 0.f, 0.f, 0.f, 0.f};
    v8f acc1 = {0.f, 0.f, 0.f, 0.f, 0.f, 0.f, 0.f, 0.f};

    for (int k0 = 0; k0 < K; k0 += 32) {
        v8bf alo = *(const v8bf*)&As[arow][k0 + kgA];
        v8bf ahi = *(const v8bf*)&As[arow][k0 + kgA + 8];
        v16bf af = __builtin_shufflevector(alo, ahi, 0, 1, 2, 3, 4, 5, 6, 7,
                                           8, 9, 10, 11, 12, 13, 14, 15);
        v8bf b0l = *(const v8bf*)&Bt[bc0][k0 + kgB];
        v8bf b0h = *(const v8bf*)&Bt[bc0][k0 + kgB + 8];
        v16bf bf0 = __builtin_shufflevector(b0l, b0h, 0, 1, 2, 3, 4, 5, 6, 7,
                                            8, 9, 10, 11, 12, 13, 14, 15);
        v8bf b1l = *(const v8bf*)&Bt[bc1][k0 + kgB];
        v8bf b1h = *(const v8bf*)&Bt[bc1][k0 + kgB + 8];
        v16bf bf1 = __builtin_shufflevector(b1l, b1h, 0, 1, 2, 3, 4, 5, 6, 7,
                                            8, 9, 10, 11, 12, 13, 14, 15);
        acc0 = __builtin_amdgcn_wmma_f32_16x16x32_bf16(false, af, false, bf0,
                                                       (short)0, acc0, false, false);
        acc1 = __builtin_amdgcn_wmma_f32_16x16x32_bf16(false, af, false, bf1,
                                                       (short)0, acc1, false, false);
    }

    const int gr_base = row_base + wr * 16 + lhi * 8;   // C layout: m = (lane>>4)*8 + r
    v8f accs[2] = {acc0, acc1};
#pragma unroll
    for (int tc = 0; tc < 2; ++tc) {
        int gcol = col_base + wcg * 32 + tc * 16 + l15;
        float bv = (bias != nullptr && gcol < Ncol) ? bias[gcol] : 0.0f;
#pragma unroll
        for (int r = 0; r < 8; ++r) {
            int grow = gr_base + r;
            if (grow < M && gcol < Ncol) {
                float v = accs[tc][r] + bv;
                if (act == ACT_SIGMOID)   v = 1.0f / (1.0f + expf(-v));
                else if (act == ACT_TANH) v = tanhf(v);
                if (outF) outF[(size_t)grow * ldo + gcol] = v;
                else      outH[(size_t)grow * ldo + gcol] = (__bf16)v;
            }
        }
    }
}

// ---------------- GCN aggregation ----------------

// self-loop term: acc[n][j] = dinv[n]^2 * xw[n][j]
__global__ void gcn_init_kernel(const float* __restrict__ xw, const float* __restrict__ dinv,
                                float* __restrict__ acc, int Nn) {
    int i = blockIdx.x * blockDim.x + threadIdx.x;
    if (i >= Nn * 64) return;
    float dv = dinv[i >> 6];
    acc[i] = dv * dv * xw[i];
}

// 16 lanes per edge, 4 feats each: acc[dst] += norm[e] * xw[src]
__global__ void gcn_scatter_kernel(const float* __restrict__ xw, const float* __restrict__ nrm,
                                   const int* __restrict__ src, const int* __restrict__ dst,
                                   float* __restrict__ acc, int E) {
    int i = blockIdx.x * blockDim.x + threadIdx.x;
    int e = i >> 4;
    if (e >= E) return;
    int f0 = (i & 15) * 4;
    int s = src[e], d = dst[e];
    float nw = nrm[e];
    const float* xs = xw + (size_t)s * 64 + f0;
    float* ad = acc + (size_t)d * 64 + f0;
#pragma unroll
    for (int j = 0; j < 4; ++j) atomicAdd(&ad[j], nw * xs[j]);
}

__global__ void sigmoid_bias_kernel(float* __restrict__ a, const float* __restrict__ b, int total) {
    int i = blockIdx.x * blockDim.x + threadIdx.x;
    if (i >= total) return;
    float v = a[i] + b[i & 63];
    a[i] = 1.0f / (1.0f + expf(-v));
}

// zu = [xi(bf16) | g | h]  (K = Fin + 128)
__global__ void build_zu_kernel(const __bf16* __restrict__ xi, const float* __restrict__ g,
                                const float* __restrict__ h, __bf16* __restrict__ zu,
                                int Nn, int Fin) {
    int K = Fin + 128;
    int i = blockIdx.x * blockDim.x + threadIdx.x;
    if (i >= Nn * K) return;
    int n = i / K, c = i - n * K;
    __bf16 v;
    if (c < Fin)            v = xi[(size_t)n * Fin + c];
    else if (c < Fin + 64)  v = (__bf16)g[(size_t)n * 64 + (c - Fin)];
    else                    v = (__bf16)h[(size_t)n * 64 + (c - Fin - 64)];
    zu[i] = v;
}

// overwrite trailing h-block of zu with r*h  (r = ur cols 64..127)
__global__ void build_zc_kernel(const float* __restrict__ ur, const float* __restrict__ h,
                                __bf16* __restrict__ zu, int Nn, int Fin) {
    int i = blockIdx.x * blockDim.x + threadIdx.x;
    if (i >= Nn * 64) return;
    int n = i >> 6, j = i & 63;
    float r = ur[(size_t)n * 128 + 64 + j];
    zu[(size_t)n * (Fin + 128) + Fin + 64 + j] = (__bf16)(r * h[i]);
}

// h = u*h + (1-u)*c ; optionally stash bf16 copy into h_seq slot
__global__ void h_update_kernel(const float* __restrict__ ur, const float* __restrict__ c,
                                float* __restrict__ h, __bf16* __restrict__ hs, int Nn) {
    int i = blockIdx.x * blockDim.x + threadIdx.x;
    if (i >= Nn * 64) return;
    int n = i >> 6, j = i & 63;
    float u = ur[(size_t)n * 128 + j];
    float hn = u * h[i] + (1.0f - u) * c[i];
    h[i] = hn;
    if (hs) hs[i] = (__bf16)hn;
}

// ---------------- temporal attention (T=12, heads=2, dh=32) ----------------
__global__ void attention_kernel(const __bf16* __restrict__ qkv, __bf16* __restrict__ o, int Nn) {
    int i = blockIdx.x * blockDim.x + threadIdx.x;
    int n = i >> 1, hh = i & 1;
    if (n >= Nn) return;
    const float scale = 0.17677669529663687f;  // 1/sqrt(32)
    for (int t = 0; t < 12; ++t) {
        float q[32];
        const __bf16* qp = qkv + ((size_t)t * Nn + n) * 192 + hh * 32;
#pragma unroll
        for (int d = 0; d < 32; ++d) q[d] = (float)qp[d];
        float s[12], mx = -1e30f;
        for (int ss = 0; ss < 12; ++ss) {
            const __bf16* kp = qkv + ((size_t)ss * Nn + n) * 192 + 64 + hh * 32;
            float a = 0.f;
#pragma unroll
            for (int d = 0; d < 32; ++d) a += q[d] * (float)kp[d];
            s[ss] = a * scale;
            mx = fmaxf(mx, s[ss]);
        }
        float sum = 0.f;
        for (int ss = 0; ss < 12; ++ss) { s[ss] = expf(s[ss] - mx); sum += s[ss]; }
        float inv = 1.0f / sum;
        __bf16* op = o + ((size_t)t * Nn + n) * 64 + hh * 32;
        for (int d = 0; d < 32; ++d) {
            float a = 0.f;
            for (int ss = 0; ss < 12; ++ss) {
                const __bf16* vp = qkv + ((size_t)ss * Nn + n) * 192 + 128 + hh * 32;
                a += s[ss] * (float)vp[d];
            }
            op[d] = (__bf16)(a * inv);
        }
    }
}

__global__ void mean_t_kernel(const __bf16* __restrict__ o, __bf16* __restrict__ obar, int Nn) {
    int i = blockIdx.x * blockDim.x + threadIdx.x;
    if (i >= Nn * 64) return;
    int n = i >> 6, j = i & 63;
    float s = 0.f;
#pragma unroll
    for (int t = 0; t < 12; ++t) s += (float)o[((size_t)t * Nn + n) * 64 + j];
    obar[i] = (__bf16)(s * (1.0f / 12.0f));
}

// ---------------- host orchestration ----------------

static inline char* bump(char*& p, size_t bytes) {
    char* r = p;
    p += (bytes + 255) & ~(size_t)255;
    return r;
}

static void launch_gemm(const __bf16* A, int lda, const __bf16* B, int ldb,
                        const float* bias, float* outF, __bf16* outH, int ldo,
                        int M, int K, int Ncol, int act, hipStream_t s) {
    dim3 grid(ceil_div(M, 64), ceil_div(Ncol, 64));
    gemm_bf16_kernel<<<grid, 256, 0, s>>>(A, lda, B, ldb, bias, outF, outH, ldo, M, K, Ncol, act);
}

extern "C" void kernel_launch(void* const* d_in, const int* in_sizes, int n_in,
                              void* d_out, int out_size, void* d_ws, size_t ws_size,
                              hipStream_t stream) {
    const int Nn = 20000, F = 32, T = 12, E = 320000;
    (void)in_sizes; (void)n_in; (void)out_size; (void)ws_size;

    const float* x   = (const float*)d_in[0];
    const int*   ei  = (const int*)d_in[1];
    const int*   src = ei;
    const int*   dst = ei + E;
    const float* ea  = (const float*)d_in[2];
    const float* Wg0f = (const float*)d_in[3];  const float* bg0 = (const float*)d_in[4];
    const float* Wu0f = (const float*)d_in[5];  const float* bu0 = (const float*)d_in[6];
    const float* Wr0f = (const float*)d_in[7];  const float* br0 = (const float*)d_in[8];
    const float* Wc0f = (const float*)d_in[9];  const float* bc0 = (const float*)d_in[10];
    const float* Wg1f = (const float*)d_in[11]; const float* bg1 = (const float*)d_in[12];
    const float* Wu1f = (const float*)d_in[13]; const float* bu1 = (const float*)d_in[14];
    const float* Wr1f = (const float*)d_in[15]; const float* br1 = (const float*)d_in[16];
    const float* Wc1f = (const float*)d_in[17]; const float* bc1 = (const float*)d_in[18];
    const float* Wqkvf = (const float*)d_in[19]; const float* bqkv = (const float*)d_in[20];
    const float* Wopf  = (const float*)d_in[21]; const float* bop  = (const float*)d_in[22];
    const float* Woutf = (const float*)d_in[23]; const float* bout = (const float*)d_in[24];

    char* p = (char*)d_ws;
    float* deg   = (float*)bump(p, (size_t)Nn * 4);
    float* dinv  = (float*)bump(p, (size_t)Nn * 4);
    float* nrm   = (float*)bump(p, (size_t)E * 4);
    __bf16* xi16 = (__bf16*)bump(p, (size_t)Nn * 64 * 2);
    float* xw    = (float*)bump(p, (size_t)Nn * 64 * 4);
    float* gacc  = (float*)bump(p, (size_t)Nn * 64 * 4);
    __bf16* zu   = (__bf16*)bump(p, (size_t)Nn * 192 * 2);
    float* urf   = (float*)bump(p, (size_t)Nn * 128 * 4);
    float* cf    = (float*)bump(p, (size_t)Nn * 64 * 4);
    float* h0f   = (float*)bump(p, (size_t)Nn * 64 * 4);
    float* h1f   = (float*)bump(p, (size_t)Nn * 64 * 4);
    __bf16* hseq = (__bf16*)bump(p, (size_t)T * Nn * 64 * 2);
    __bf16* qkv16 = (__bf16*)bump(p, (size_t)T * Nn * 192 * 2);
    __bf16* o16   = (__bf16*)bump(p, (size_t)T * Nn * 64 * 2);
    __bf16* obar16 = (__bf16*)bump(p, (size_t)Nn * 64 * 2);
    __bf16* hattn16 = (__bf16*)bump(p, (size_t)Nn * 64 * 2);
    __bf16* wg0  = (__bf16*)bump(p, 32 * 64 * 2);
    __bf16* wur0 = (__bf16*)bump(p, 160 * 128 * 2);
    __bf16* wc0  = (__bf16*)bump(p, 160 * 64 * 2);
    __bf16* wg1  = (__bf16*)bump(p, 64 * 64 * 2);
    __bf16* wur1 = (__bf16*)bump(p, 192 * 128 * 2);
    __bf16* wc1  = (__bf16*)bump(p, 192 * 64 * 2);
    __bf16* wqkv = (__bf16*)bump(p, 64 * 192 * 2);
    __bf16* wop  = (__bf16*)bump(p, 64 * 64 * 2);
    __bf16* wout = (__bf16*)bump(p, 64 * 12 * 2);
    float* bur0  = (float*)bump(p, 128 * 4);
    float* bur1  = (float*)bump(p, 128 * 4);

    const int TB = 256;
    hipStream_t s = stream;

    // --- graph norm ---
    fill_f32_kernel<<<ceil_div(Nn, TB), TB, 0, s>>>(deg, 1.0f, Nn);  // self-loop weight 1
    deg_scatter_kernel<<<ceil_div(E, TB), TB, 0, s>>>(dst, ea, deg, E);
    dinv_kernel<<<ceil_div(Nn, TB), TB, 0, s>>>(deg, dinv, Nn);
    norm_kernel<<<ceil_div(E, TB), TB, 0, s>>>(src, dst, ea, dinv, nrm, E);

    // --- pack weights to bf16 (Wu|Wr fused; qkv & out_proj transposed) ---
    pack_w_kernel<<<ceil_div(32 * 64, TB), TB, 0, s>>>(Wg0f, wg0, 32, 64, 64, 0, 0);
    pack_w_kernel<<<ceil_div(160 * 64, TB), TB, 0, s>>>(Wu0f, wur0, 160, 64, 128, 0, 0);
    pack_w_kernel<<<ceil_div(160 * 64, TB), TB, 0, s>>>(Wr0f, wur0, 160, 64, 128, 64, 0);
    pack_w_kernel<<<ceil_div(160 * 64, TB), TB, 0, s>>>(Wc0f, wc0, 160, 64, 64, 0, 0);
    pack_w_kernel<<<ceil_div(64 * 64, TB), TB, 0, s>>>(Wg1f, wg1, 64, 64, 64, 0, 0);
    pack_w_kernel<<<ceil_div(192 * 64, TB), TB, 0, s>>>(Wu1f, wur1, 192, 64, 128, 0, 0);
    pack_w_kernel<<<ceil_div(192 * 64, TB), TB, 0, s>>>(Wr1f, wur1, 192, 64, 128, 64, 0);
    pack_w_kernel<<<ceil_div(192 * 64, TB), TB, 0, s>>>(Wc1f, wc1, 192, 64, 64, 0, 0);
    pack_w_kernel<<<ceil_div(64 * 192, TB), TB, 0, s>>>(Wqkvf, wqkv, 64, 192, 192, 0, 1);
    pack_w_kernel<<<ceil_div(64 * 64, TB), TB, 0, s>>>(Wopf, wop, 64, 64, 64, 0, 1);
    pack_w_kernel<<<ceil_div(64 * 12, TB), TB, 0, s>>>(Woutf, wout, 64, 12, 12, 0, 0);
    copy_bias_kernel<<<1, 64, 0, s>>>(bu0, bur0, 64, 0);
    copy_bias_kernel<<<1, 64, 0, s>>>(br0, bur0, 64, 64);
    copy_bias_kernel<<<1, 64, 0, s>>>(bu1, bur1, 64, 0);
    copy_bias_kernel<<<1, 64, 0, s>>>(br1, bur1, 64, 64);

    fill_f32_kernel<<<ceil_div(Nn * 64, TB), TB, 0, s>>>(h0f, 0.0f, Nn * 64);
    fill_f32_kernel<<<ceil_div(Nn * 64, TB), TB, 0, s>>>(h1f, 0.0f, Nn * 64);

    const int NE = Nn * 64;
    for (int t = 0; t < T; ++t) {
        // ---- cell 0: xi = x[:, :, t] (F=32), h = h0 ----
        pack_xi_kernel<<<ceil_div(Nn * 32, TB), TB, 0, s>>>(x, xi16, Nn, 32, F * T, T, t);
        launch_gemm(xi16, 32, wg0, 64, nullptr, xw, nullptr, 64, Nn, 32, 64, ACT_NONE, s);
        gcn_init_kernel<<<ceil_div(NE, TB), TB, 0, s>>>(xw, dinv, gacc, Nn);
        gcn_scatter_kernel<<<ceil_div(E * 16, TB), TB, 0, s>>>(xw, nrm, src, dst, gacc, E);
        sigmoid_bias_kernel<<<ceil_div(NE, TB), TB, 0, s>>>(gacc, bg0, NE);
        build_zu_kernel<<<ceil_div(Nn * 160, TB), TB, 0, s>>>(xi16, gacc, h0f, zu, Nn, 32);
        launch_gemm(zu, 160, wur0, 128, bur0, urf, nullptr, 128, Nn, 160, 128, ACT_SIGMOID, s);
        build_zc_kernel<<<ceil_div(NE, TB), TB, 0, s>>>(urf, h0f, zu, Nn, 32);
        launch_gemm(zu, 160, wc0, 64, bc0, cf, nullptr, 64, Nn, 160, 64, ACT_TANH, s);
        h_update_kernel<<<ceil_div(NE, TB), TB, 0, s>>>(urf, cf, h0f, nullptr, Nn);

        // ---- cell 1: xi = h0 (64), h = h1 ----
        pack_xi_kernel<<<ceil_div(Nn * 64, TB), TB, 0, s>>>(h0f, xi16, Nn, 64, 64, 1, 0);
        launch_gemm(xi16, 64, wg1, 64, nullptr, xw, nullptr, 64, Nn, 64, 64, ACT_NONE, s);
        gcn_init_kernel<<<ceil_div(NE, TB), TB, 0, s>>>(xw, dinv, gacc, Nn);
        gcn_scatter_kernel<<<ceil_div(E * 16, TB), TB, 0, s>>>(xw, nrm, src, dst, gacc, E);
        sigmoid_bias_kernel<<<ceil_div(NE, TB), TB, 0, s>>>(gacc, bg1, NE);
        build_zu_kernel<<<ceil_div(Nn * 192, TB), TB, 0, s>>>(xi16, gacc, h1f, zu, Nn, 64);
        launch_gemm(zu, 192, wur1, 128, bur1, urf, nullptr, 128, Nn, 192, 128, ACT_SIGMOID, s);
        build_zc_kernel<<<ceil_div(NE, TB), TB, 0, s>>>(urf, h1f, zu, Nn, 64);
        launch_gemm(zu, 192, wc1, 64, bc1, cf, nullptr, 64, Nn, 192, 64, ACT_TANH, s);
        h_update_kernel<<<ceil_div(NE, TB), TB, 0, s>>>(urf, cf, h1f, hseq + (size_t)t * NE, Nn);
    }

    // ---- attention over time ----
    launch_gemm(hseq, 64, wqkv, 192, bqkv, nullptr, qkv16, 192, T * Nn, 64, 192, ACT_NONE, s);
    attention_kernel<<<ceil_div(Nn * 2, TB), TB, 0, s>>>(qkv16, o16, Nn);
    mean_t_kernel<<<ceil_div(NE, TB), TB, 0, s>>>(o16, obar16, Nn);
    // mean commutes with the linear out-projection: h_attn = (mean_t o) @ Wop^T + b
    launch_gemm(obar16, 64, wop, 64, bop, nullptr, hattn16, 64, Nn, 64, 64, ACT_NONE, s);
    launch_gemm(hattn16, 64, wout, 12, bout, (float*)d_out, nullptr, 12, Nn, 64, 12, ACT_NONE, s);
}
